// MSDeformAttnTransformerEncoderOnly_14851996910122
// MI455X (gfx1250) — matrix operand order repaired and checked by reading, not compiled, not run on hardware
//
#include <hip/hip_runtime.h>
#include <hip/hip_bf16.h>

#define LEN    5440
#define BATCHN 2
#define TOK    (BATCHN * LEN)   // 10880 = 85 * 128
#define DM     256
#define DFFN   1024

typedef __bf16 bf16_t;
typedef __attribute__((ext_vector_type(16))) __bf16 v16bf;
typedef __attribute__((ext_vector_type(8)))  __bf16 v8bf;
typedef __attribute__((ext_vector_type(8)))  float  v8f;

#if __has_builtin(__builtin_amdgcn_global_load_async_to_lds_b128)
#define HAVE_ASYNC_LDS 1
#else
#define HAVE_ASYNC_LDS 0
#endif

__device__ __forceinline__ bf16_t f2b(float f) {
    union { float f; unsigned u; } v; v.f = f;
    unsigned r = v.u + 0x7FFFu + ((v.u >> 16) & 1u);   // RNE
    unsigned short h = (unsigned short)(r >> 16);
    return __builtin_bit_cast(bf16_t, h);
}

// 16-byte global -> LDS copy, async (ASYNCcnt) when available
__device__ __forceinline__ void async_cp16(const bf16_t* g, bf16_t* l)
{
#if HAVE_ASYNC_LDS
    typedef int v4i __attribute__((vector_size(16)));
    typedef __attribute__((address_space(1))) v4i* gp_t;
    typedef __attribute__((address_space(3))) v4i* lp_t;
    __builtin_amdgcn_global_load_async_to_lds_b128((gp_t)g, (lp_t)l, 0, 0);
#else
    *(v8bf*)l = *(const v8bf*)g;
#endif
}

__device__ __forceinline__ void wait_async()
{
#if HAVE_ASYNC_LDS
#if __has_builtin(__builtin_amdgcn_s_wait_asynccnt)
    __builtin_amdgcn_s_wait_asynccnt(0);
#else
    asm volatile("s_wait_asynccnt 0x0" ::: "memory");
#endif
#endif
}

// ---------------------------------------------------------------------------
// Flatten NCHW sources -> (N, Len, C); pos gets level_embed added.
// ---------------------------------------------------------------------------
__global__ void flatten_kernel(const float* __restrict__ s0, const float* __restrict__ s1,
                               const float* __restrict__ s2, const float* __restrict__ s3,
                               const float* __restrict__ p0, const float* __restrict__ p1,
                               const float* __restrict__ p2, const float* __restrict__ p3,
                               const float* __restrict__ lvl,
                               float* __restrict__ x, float* __restrict__ pos)
{
    int i = blockIdx.x * 256 + threadIdx.x;
    if (i >= TOK * DM) return;
    int c = i & 255;
    int q = (i >> 8) % LEN;
    int n = i / (LEN * DM);
    const float *sp, *pp; int l, HW, idx;
    if (q < 4096)      { l = 0; HW = 4096; idx = q;        sp = s0; pp = p0; }
    else if (q < 5120) { l = 1; HW = 1024; idx = q - 4096; sp = s1; pp = p1; }
    else if (q < 5376) { l = 2; HW = 256;  idx = q - 5120; sp = s2; pp = p2; }
    else               { l = 3; HW = 64;   idx = q - 5376; sp = s3; pp = p3; }
    size_t si = (size_t)(n * DM + c) * HW + idx;
    x[i]   = sp[si];
    pos[i] = pp[si] + lvl[l * DM + c];
}

__global__ void unflatten_kernel(const float* __restrict__ x, float* __restrict__ out)
{
    int i = blockIdx.x * 256 + threadIdx.x;
    if (i >= TOK * DM) return;
    int c = i & 255;
    int q = (i >> 8) % LEN;
    int n = i / (LEN * DM);
    int HW, idx; size_t base;
    if (q < 4096)      { HW = 4096; idx = q;        base = 0; }
    else if (q < 5120) { HW = 1024; idx = q - 4096; base = (size_t)BATCHN * DM * 4096; }
    else if (q < 5376) { HW = 256;  idx = q - 5120; base = (size_t)BATCHN * DM * (4096 + 1024); }
    else               { HW = 64;   idx = q - 5376; base = (size_t)BATCHN * DM * (4096 + 1024 + 256); }
    out[base + (size_t)(n * DM + c) * HW + idx] = x[i];
}

// ---------------------------------------------------------------------------
// Elementwise converts
// ---------------------------------------------------------------------------
__global__ void f2bf_kernel(const float* __restrict__ in, bf16_t* __restrict__ out, int n)
{
    int i = blockIdx.x * 256 + threadIdx.x;
    if (i < n) out[i] = f2b(in[i]);
}

__global__ void make_q_kernel(const float* __restrict__ x, const float* __restrict__ pos,
                              bf16_t* __restrict__ qb, bf16_t* __restrict__ xb, int n)
{
    int i = blockIdx.x * 256 + threadIdx.x;
    if (i < n) { float xv = x[i]; qb[i] = f2b(xv + pos[i]); xb[i] = f2b(xv); }
}

// Transpose + convert weights: W[L][K][N] fp32 -> Wt[L][N][K] bf16
__global__ void wtrans_kernel(const float* __restrict__ W, bf16_t* __restrict__ Wt,
                              int K, int N, int total)
{
    int i = blockIdx.x * 256 + threadIdx.x;
    if (i >= total) return;
    int per = K * N;
    int l = i / per, r = i % per;
    int k = r / N, n = r % N;
    Wt[(size_t)l * per + (size_t)n * K + k] = f2b(W[i]);
}

// ---------------------------------------------------------------------------
// WMMA GEMM:  C[M,N] = A[M,K](bf16) * Wt[N,K]^T(bf16) + bias[N]
// Block = 256 threads (8 waves). Tile 128x64, wave tile 32x32 (4 WMMAs/step),
// K step 32, double-buffered LDS filled with async global->LDS copies.
// ---------------------------------------------------------------------------
template <bool RELU, bool OUT_BF16>
__global__ __launch_bounds__(256)
void wmma_gemm(const bf16_t* __restrict__ A, const bf16_t* __restrict__ Wt,
               const float* __restrict__ bias, void* __restrict__ Cout,
               int N, int K)
{
    __shared__ __attribute__((aligned(32))) bf16_t lds_a[2][128 * 32];  // [m][k]
    __shared__ __attribute__((aligned(32))) bf16_t lds_b[2][64 * 32];   // [n][k]

    const int tid  = threadIdx.x;
    const int wave = tid >> 5;
    const int lane = tid & 31;
    const int m0 = blockIdx.x * 128;
    const int n0 = blockIdx.y * 64;
    const int wm = (wave & 3) * 32;       // wave row base within block tile
    const int wc = (wave >> 2) * 32;      // wave col base within block tile

    const int srow = tid >> 2;            // 0..63
    const int scol = (tid & 3) * 8;       // 0..24

    const int frow = lane & 15;
    const int fka  = (lane >> 4) * 8;     // A K-half base
    const int fkb  = (lane >> 4) * 8;     // B K-half base

    v8f acc00 = {}, acc01 = {}, acc10 = {}, acc11 = {};

    const int steps = K >> 5;

    auto stage = [&](int buf, int k0) {
        const bf16_t* ag = A + (size_t)(m0 + srow) * K + k0 + scol;
        async_cp16(ag,                  &lds_a[buf][srow * 32 + scol]);
        async_cp16(ag + (size_t)64 * K, &lds_a[buf][(srow + 64) * 32 + scol]);
        const bf16_t* bg = Wt + (size_t)(n0 + srow) * K + k0 + scol;
        async_cp16(bg,                  &lds_b[buf][srow * 32 + scol]);
    };

    stage(0, 0);
    for (int s = 0; s < steps; ++s) {
        wait_async();          // our async fills of buf[s&1] are done
        __syncthreads();       // everyone's fills done; prior reads of buf[(s+1)&1] done
        if (s + 1 < steps) stage((s + 1) & 1, (s + 1) << 5);

        const bf16_t* la = lds_a[s & 1];
        const bf16_t* lb = lds_b[s & 1];

        // A fragments: lane holds row, K = {fka..fka+7, fka+16..fka+23}
        v16bf a0, a1;
        {
            const bf16_t* p = la + (wm + frow) * 32;
            v8bf lo = *(const v8bf*)(p + fka);
            v8bf hi = *(const v8bf*)(p + fka + 16);
#pragma unroll
            for (int i = 0; i < 8; ++i) { a0[i] = lo[i]; a0[8 + i] = hi[i]; }
            p += 16 * 32;
            lo = *(const v8bf*)(p + fka);
            hi = *(const v8bf*)(p + fka + 16);
#pragma unroll
            for (int i = 0; i < 8; ++i) { a1[i] = lo[i]; a1[8 + i] = hi[i]; }
        }
        // B fragments: lane holds col, K = {fkb..fkb+7, fkb+16..fkb+23}
        v16bf b0, b1;
        {
            const bf16_t* p = lb + (wc + frow) * 32;
            v8bf lo = *(const v8bf*)(p + fkb);
            v8bf hi = *(const v8bf*)(p + fkb + 16);
#pragma unroll
            for (int i = 0; i < 8; ++i) { b0[i] = lo[i]; b0[8 + i] = hi[i]; }
            p += 16 * 32;
            lo = *(const v8bf*)(p + fkb);
            hi = *(const v8bf*)(p + fkb + 16);
#pragma unroll
            for (int i = 0; i < 8; ++i) { b1[i] = lo[i]; b1[8 + i] = hi[i]; }
        }

        acc00 = __builtin_amdgcn_wmma_f32_16x16x32_bf16(false, a0, false, b0, (short)0, acc00, false, false);
        acc01 = __builtin_amdgcn_wmma_f32_16x16x32_bf16(false, a0, false, b1, (short)0, acc01, false, false);
        acc10 = __builtin_amdgcn_wmma_f32_16x16x32_bf16(false, a1, false, b0, (short)0, acc10, false, false);
        acc11 = __builtin_amdgcn_wmma_f32_16x16x32_bf16(false, a1, false, b1, (short)0, acc11, false, false);
    }

    // epilogue: lane -> col = lane&15, rows = (lane>>4)*8 + r (per 16x16 tile)
    const int col = lane & 15;
    const int rb  = (lane >> 4) * 8;
#pragma unroll
    for (int i = 0; i < 2; ++i) {
        v8f ar0 = i ? acc10 : acc00;
        v8f ar1 = i ? acc11 : acc01;
#pragma unroll
        for (int j = 0; j < 2; ++j) {
            v8f acc = j ? ar1 : ar0;
            const int cg = n0 + wc + j * 16 + col;
            const float bsv = bias[cg];
#pragma unroll
            for (int r = 0; r < 8; ++r) {
                float vv = acc[r] + bsv;
                if (RELU) vv = fmaxf(vv, 0.0f);
                size_t oi = (size_t)(m0 + wm + i * 16 + rb + r) * N + cg;
                if (OUT_BF16) ((bf16_t*)Cout)[oi] = f2b(vv);
                else          ((float*)Cout)[oi]  = vv;
            }
        }
    }
}

// ---------------------------------------------------------------------------
// Softmax over 16 attention-weight logits per (token, head)
// ---------------------------------------------------------------------------
__global__ void softmax16_kernel(float* __restrict__ aw, int rows)
{
    int r = blockIdx.x * 256 + threadIdx.x;
    if (r >= rows) return;
    float* p = aw + (size_t)r * 16;
    float m = p[0];
#pragma unroll
    for (int i = 1; i < 16; ++i) m = fmaxf(m, p[i]);
    float s = 0.f;
#pragma unroll
    for (int i = 0; i < 16; ++i) { float e = __expf(p[i] - m); p[i] = e; s += e; }
    float inv = 1.0f / s;
#pragma unroll
    for (int i = 0; i < 16; ++i) p[i] *= inv;
}

// ---------------------------------------------------------------------------
// Deformable attention sampling: one wave32 per (batch, query, head); lane = dim
// ---------------------------------------------------------------------------
__global__ __launch_bounds__(256)
void deform_sample_kernel(const float* __restrict__ val, const float* __restrict__ offs,
                          const float* __restrict__ aw, float* __restrict__ out)
{
    const int wid  = blockIdx.x * 8 + (threadIdx.x >> 5);
    const int lane = threadIdx.x & 31;
    const int head = wid & 7;
    const int q    = (wid >> 3) % LEN;
    const int b    = wid / (8 * LEN);

    int Wq, Hq, idx;
    if (q < 4096)      { Hq = 64; Wq = 64; idx = q; }
    else if (q < 5120) { Hq = 32; Wq = 32; idx = q - 4096; }
    else if (q < 5376) { Hq = 16; Wq = 16; idx = q - 5120; }
    else               { Hq = 8;  Wq = 8;  idx = q - 5376; }
    const float rx = ((idx % Wq) + 0.5f) / (float)Wq;
    const float ry = ((idx / Wq) + 0.5f) / (float)Hq;

    const size_t tokbase = (size_t)(b * LEN + q);
    const float* op = offs + tokbase * 256 + head * 32;
    const float* ap = aw   + tokbase * 128 + head * 16;
    const float* vb = val  + (size_t)b * LEN * 256 + head * 32 + lane;

    const int Hs[4] = {64, 32, 16, 8};
    const int St[4] = {0, 4096, 5120, 5376};

    float acc = 0.f;
#pragma unroll
    for (int l = 0; l < 4; ++l) {
        const int Hl = Hs[l], Wl = Hs[l], s0 = St[l];
#pragma unroll
        for (int p = 0; p < 4; ++p) {
            const float ox = op[(l * 4 + p) * 2 + 0];
            const float oy = op[(l * 4 + p) * 2 + 1];
            const float w  = ap[l * 4 + p];
            const float xx = (rx + ox / (float)Wl) * (float)Wl - 0.5f;
            const float yy = (ry + oy / (float)Hl) * (float)Hl - 0.5f;
            const float xf = floorf(xx), yf = floorf(yy);
            const float wx = xx - xf,   wy = yy - yf;
            const int   x0 = (int)xf,   y0 = (int)yf;
#pragma unroll
            for (int dy = 0; dy < 2; ++dy) {
#pragma unroll
                for (int dx = 0; dx < 2; ++dx) {
                    int xi = x0 + dx, yi = y0 + dy;
                    float bw = (dx ? wx : 1.f - wx) * (dy ? wy : 1.f - wy);
                    if (xi >= 0 && xi < Wl && yi >= 0 && yi < Hl) {
                        int tok = s0 + yi * Wl + xi;
                        acc += w * bw * vb[(size_t)tok * 256];
                    }
                }
            }
        }
    }
    out[tokbase * 256 + head * 32 + lane] = acc;
}

// ---------------------------------------------------------------------------
// x = LayerNorm(x + t) * g + b ; one wave32 per token, lane handles 8 channels
// ---------------------------------------------------------------------------
__global__ __launch_bounds__(256)
void add_ln_kernel(float* __restrict__ x, const float* __restrict__ t,
                   const float* __restrict__ g, const float* __restrict__ b)
{
    const int tok  = blockIdx.x * 8 + (threadIdx.x >> 5);
    const int lane = threadIdx.x & 31;
    float* xr = x + (size_t)tok * DM;
    const float* tr = t + (size_t)tok * DM;
    float v[8];
    float s = 0.f;
#pragma unroll
    for (int i = 0; i < 8; ++i) { v[i] = xr[lane + i * 32] + tr[lane + i * 32]; s += v[i]; }
#pragma unroll
    for (int off = 16; off; off >>= 1) s += __shfl_xor(s, off, 32);
    const float mean = s * (1.0f / DM);
    float vs = 0.f;
#pragma unroll
    for (int i = 0; i < 8; ++i) { float d = v[i] - mean; vs += d * d; }
#pragma unroll
    for (int off = 16; off; off >>= 1) vs += __shfl_xor(vs, off, 32);
    const float inv = rsqrtf(vs * (1.0f / DM) + 1e-5f);
#pragma unroll
    for (int i = 0; i < 8; ++i) {
        int c = lane + i * 32;
        xr[c] = (v[i] - mean) * inv * g[c] + b[c];
    }
}

// ---------------------------------------------------------------------------
extern "C" void kernel_launch(void* const* d_in, const int* in_sizes, int n_in,
                              void* d_out, int out_size, void* d_ws, size_t ws_size,
                              hipStream_t stream)
{
    (void)in_sizes; (void)n_in; (void)out_size; (void)ws_size;

    const float* src[4] = {(const float*)d_in[0], (const float*)d_in[1],
                           (const float*)d_in[2], (const float*)d_in[3]};
    const float* pin[4] = {(const float*)d_in[4], (const float*)d_in[5],
                           (const float*)d_in[6], (const float*)d_in[7]};
    const float* lvl = (const float*)d_in[8];
    const float* Ws  = (const float*)d_in[9];   const float* bs  = (const float*)d_in[10];
    const float* Wa  = (const float*)d_in[11];  const float* ba  = (const float*)d_in[12];
    const float* Wv  = (const float*)d_in[13];  const float* bvp = (const float*)d_in[14];
    const float* Wo  = (const float*)d_in[15];  const float* bo  = (const float*)d_in[16];
    const float* g1  = (const float*)d_in[17];  const float* b1  = (const float*)d_in[18];
    const float* Wf1 = (const float*)d_in[19];  const float* bf1 = (const float*)d_in[20];
    const float* Wf2 = (const float*)d_in[21];  const float* bf2 = (const float*)d_in[22];
    const float* g2  = (const float*)d_in[23];  const float* b2  = (const float*)d_in[24];

    char* w = (char*)d_ws;
    auto alloc = [&](size_t bytes) { char* p = w; w += (bytes + 255) & ~(size_t)255; return p; };

    float*  x    = (float*)alloc((size_t)TOK * DM * 4);
    float*  posf = (float*)alloc((size_t)TOK * DM * 4);
    float*  tmp  = (float*)alloc((size_t)TOK * DM * 4);   // offs / proj / ff
    float*  valb = (float*)alloc((size_t)TOK * DM * 4);
    float*  attn = (float*)alloc((size_t)TOK * DM * 4);
    float*  awb  = (float*)alloc((size_t)TOK * 128 * 4);
    bf16_t* qb   = (bf16_t*)alloc((size_t)TOK * DM * 2);
    bf16_t* xb   = (bf16_t*)alloc((size_t)TOK * DM * 2);
    bf16_t* atb  = (bf16_t*)alloc((size_t)TOK * DM * 2);
    bf16_t* hidb = (bf16_t*)alloc((size_t)TOK * DFFN * 2);
    bf16_t* ws_b = (bf16_t*)alloc((size_t)6 * 256 * 256 * 2);
    bf16_t* wa_b = (bf16_t*)alloc((size_t)6 * 256 * 128 * 2);
    bf16_t* wv_b = (bf16_t*)alloc((size_t)6 * 256 * 256 * 2);
    bf16_t* wo_b = (bf16_t*)alloc((size_t)6 * 256 * 256 * 2);
    bf16_t* w1_b = (bf16_t*)alloc((size_t)6 * 256 * 1024 * 2);
    bf16_t* w2_b = (bf16_t*)alloc((size_t)6 * 1024 * 256 * 2);

    const int NE = TOK * DM;                 // 2,785,280
    const int EB = (NE + 255) / 256;

    flatten_kernel<<<EB, 256, 0, stream>>>(src[0], src[1], src[2], src[3],
                                           pin[0], pin[1], pin[2], pin[3],
                                           lvl, x, posf);

    // weights -> bf16, transposed to [N][K] (all layers at once)
    wtrans_kernel<<<(6*256*256 + 255)/256, 256, 0, stream>>>(Ws,  ws_b, 256, 256,  6*256*256);
    wtrans_kernel<<<(6*256*128 + 255)/256, 256, 0, stream>>>(Wa,  wa_b, 256, 128,  6*256*128);
    wtrans_kernel<<<(6*256*256 + 255)/256, 256, 0, stream>>>(Wv,  wv_b, 256, 256,  6*256*256);
    wtrans_kernel<<<(6*256*256 + 255)/256, 256, 0, stream>>>(Wo,  wo_b, 256, 256,  6*256*256);
    wtrans_kernel<<<(6*256*1024 + 255)/256, 256, 0, stream>>>(Wf1, w1_b, 256, 1024, 6*256*1024);
    wtrans_kernel<<<(6*1024*256 + 255)/256, 256, 0, stream>>>(Wf2, w2_b, 1024, 256, 6*1024*256);

    const dim3 blk(256);
    const int MB = TOK / 128;                // 85 M-tiles

    for (int l = 0; l < 6; ++l) {
        const bf16_t* wsl = ws_b + (size_t)l * 256 * 256;
        const bf16_t* wal = wa_b + (size_t)l * 256 * 128;
        const bf16_t* wvl = wv_b + (size_t)l * 256 * 256;
        const bf16_t* wol = wo_b + (size_t)l * 256 * 256;
        const bf16_t* w1l = w1_b + (size_t)l * 256 * 1024;
        const bf16_t* w2l = w2_b + (size_t)l * 1024 * 256;
        const float* bsl = bs  + l * 256;
        const float* bal = ba  + l * 128;
        const float* bvl = bvp + l * 256;
        const float* bol = bo  + l * 256;
        const float* b1l = bf1 + l * 1024;
        const float* b2l = bf2 + l * 256;

        make_q_kernel<<<EB, blk, 0, stream>>>(x, posf, qb, xb, NE);

        wmma_gemm<false, false><<<dim3(MB, 4),  blk, 0, stream>>>(qb,   wsl, bsl, tmp,  256, 256);
        wmma_gemm<false, false><<<dim3(MB, 2),  blk, 0, stream>>>(qb,   wal, bal, awb,  128, 256);
        softmax16_kernel<<<(TOK * 8) / 256, blk, 0, stream>>>(awb, TOK * 8);
        wmma_gemm<false, false><<<dim3(MB, 4),  blk, 0, stream>>>(xb,   wvl, bvl, valb, 256, 256);

        deform_sample_kernel<<<TOK, blk, 0, stream>>>(valb, tmp, awb, attn);

        f2bf_kernel<<<EB, blk, 0, stream>>>(attn, atb, NE);
        wmma_gemm<false, false><<<dim3(MB, 4),  blk, 0, stream>>>(atb,  wol, bol, tmp,  256, 256);
        add_ln_kernel<<<TOK / 8, blk, 0, stream>>>(x, tmp, g1 + l * 256, b1 + l * 256);

        f2bf_kernel<<<EB, blk, 0, stream>>>(x, xb, NE);
        wmma_gemm<true,  true ><<<dim3(MB, 16), blk, 0, stream>>>(xb,   w1l, b1l, hidb, 1024, 256);
        wmma_gemm<false, false><<<dim3(MB, 4),  blk, 0, stream>>>(hidb, w2l, b2l, tmp,  256, 1024);
        add_ln_kernel<<<TOK / 8, blk, 0, stream>>>(x, tmp, g2 + l * 256, b2 + l * 256);
    }

    unflatten_kernel<<<EB, blk, 0, stream>>>(x, (float*)d_out);
}
